// ASAP_58033598104040
// MI455X (gfx1250) — compile-verified, matrix operand order checked by reading, not run on hardware
//
#include <hip/hip_runtime.h>
#include <math.h>

// ---------------------------------------------------------------------------
// Problem constants (match reference)
// ---------------------------------------------------------------------------
static constexpr int N    = 50000;
static constexpr int E    = 800000;
static constexpr int B    = 8;
static constexpr int HID  = 64;
static constexpr int NCLS = 40;
static constexpr float EPS = 1e-5f;

static constexpr int EBLK = E / 64;   // 12500 blocks, 4 waves x 16 edges each

// ---------------------------------------------------------------------------
// Types for CDNA5 WMMA (wave32)
// ---------------------------------------------------------------------------
typedef __attribute__((ext_vector_type(16))) __bf16 v16bf;
typedef __attribute__((ext_vector_type(8)))  float  v8f;

union AFrag { unsigned u[8]; v16bf v; };

__device__ __forceinline__ unsigned short f2bf(float f) {
  unsigned u = __float_as_uint(f);
  unsigned r = u + 0x7FFFu + ((u >> 16) & 1u);   // round-to-nearest-even
  return (unsigned short)(r >> 16);
}

__device__ __forceinline__ void atomicMaxFloat(float* addr, float v) {
  // Sign-aware monotonic mapping: works with -inf init and mixed signs.
  if (v >= 0.f) atomicMax((int*)addr, __float_as_int(v));
  else          atomicMin((unsigned int*)addr, (unsigned int)__float_as_int(v));
}

// ---------------------------------------------------------------------------
// Weight pack: row-major [K,64] f32 -> WMMA B-fragment layout, bf16.
// Tile (kt,nt): 32 lanes x 16 bf16. Lane l (l<16: koff=0, else koff=8),
// dword q<4: K = kt*32+koff+2q ; q>=4: K = kt*32+16+koff+2(q-4); N = nt*16+(l&15)
// ---------------------------------------------------------------------------
__global__ void pack_w_kernel(const float* __restrict__ W,
                              unsigned short* __restrict__ out,
                              int K, int KPAD) {
  int idx = blockIdx.x * blockDim.x + threadIdx.x;
  int total = KPAD * 64;
  if (idx >= total) return;
  int tile = idx >> 9;        // /512
  int rem  = idx & 511;
  int lane = rem >> 4;        // 0..31
  int j    = rem & 15;        // element within lane frag
  int kt = tile >> 2, nt = tile & 3;
  int n = nt * 16 + (lane & 15);
  int koff = (lane < 16) ? 0 : 8;
  int pr = j >> 1, w2 = j & 1;
  int k = kt * 32 + ((pr < 4) ? (koff + 2 * pr) : (16 + koff + 2 * (pr - 4))) + w2;
  float v = (k < K) ? W[k * 64 + n] : 0.f;
  out[idx] = f2bf(v);
}

// ---------------------------------------------------------------------------
// Misc small kernels
// ---------------------------------------------------------------------------
__global__ void fill_kernel(float* p, float v, int n) {
  int i = blockIdx.x * blockDim.x + threadIdx.x;
  if (i < n) p[i] = v;
}

__global__ void count_kernel(const int* __restrict__ batch, float* cnt) {
  int n = blockIdx.x * blockDim.x + threadIdx.x;
  if (n < N) atomicAdd(&cnt[batch[n]], 1.0f);
}

// stats: [0..63]=sum, [64..127]=sumsq  ->  bn: [0..63]=a, [64..127]=c
__global__ void bn_finalize_kernel(const float* __restrict__ stats,
                                   const float* __restrict__ g,
                                   const float* __restrict__ be,
                                   float* __restrict__ bn, float invE) {
  int c = threadIdx.x;
  if (c >= 64) return;
  float mu  = stats[c] * invE;
  float var = stats[64 + c] * invE - mu * mu;
  float rs  = rsqrtf(var + EPS);
  float a   = g[c] * rs;
  bn[c]      = a;
  bn[64 + c] = be[c] - mu * a;
}

// agg (init -inf) -> hA = max(agg,0) ; accumulate mean-pool sums per graph
__global__ void relu_pool_kernel(const float* __restrict__ agg,
                                 float* __restrict__ hA,
                                 const int* __restrict__ batch,
                                 float* __restrict__ pooled) {
  int t = blockIdx.x * blockDim.x + threadIdx.x;
  if (t >= N * 64) return;
  int n = t >> 6, c = t & 63;
  float v = fmaxf(agg[t], 0.f);
  hA[t] = v;
  atomicAdd(&pooled[batch[n] * 64 + c], v);
}

// ---------------------------------------------------------------------------
// Fused edge-conv stage kernel.
//   STAGE 0: build msg, GEMM1, relu+bias -> accumulate BN1 stats
//   STAGE 1: + apply BN1, GEMM2, relu+bias -> accumulate BN2 stats
//   STAGE 2: + apply BN2, scatter-max into agg[dst]
// Block = 128 threads (4 waves); each wave: 16 edges x 64 out channels.
// ---------------------------------------------------------------------------
struct ConvParams {
  const float* pos;        // [N,3]
  const float* x3;         // [N,3]   (conv1 only)
  const float* hN;         // [N,64]  (convw only)
  const float* edge_attr;  // [E]     (conv1 only)
  const int*   src;
  const int*   dst;
  const unsigned short* W0p;  // packed frags
  const unsigned short* W1p;
  const float* b0;
  const float* b1;
  float*       stats;      // [256]: sum1,sq1,sum2,sq2
  const float* bn1;        // [128]: a,c
  const float* bn2;        // [128]: a,c
  float*       agg;        // [N,64]
};

template <int KPAD, bool IS_CONV1, int STAGE>
__global__ __launch_bounds__(128)
void conv_stage_kernel(ConvParams p) {
  __shared__ unsigned short lds_m[4][16][KPAD];   // per-wave message tile (bf16 bits)
  __shared__ unsigned short lds_m2[4][16][64];    // per-wave BN1(h1) tile
  __shared__ float lds_stats[128];

  const int tid  = threadIdx.x;
  const int w    = tid >> 5;
  const int lane = tid & 31;
  const int eb   = (blockIdx.x * 4 + w) * 16;     // wave's base edge

  if (STAGE != 2) lds_stats[tid] = 0.f;

  // ---- build messages: two lanes per edge, each covers half of KPAD ----
  {
    const int i = lane >> 1;
    const int e = eb + i;
    const int s = p.src[e];
    const int t = p.dst[e];
    const int k0 = (lane & 1) * (KPAD / 2);
    #pragma unroll 4
    for (int kk = 0; kk < KPAD / 2; ++kk) {
      const int k = k0 + kk;
      float v;
      if (IS_CONV1) {
        if (k == 0) {
          v = p.edge_attr[e];
        } else if (k < 7) {
          int j = k - 1;
          float vs = (j < 3) ? p.pos[s * 3 + j] : p.x3[s * 3 + j - 3];
          float vt = (j < 3) ? p.pos[t * 3 + j] : p.x3[t * 3 + j - 3];
          v = vs - vt;
        } else if (k < 13) {
          int j = k - 7;
          v = (j < 3) ? p.pos[t * 3 + j] : p.x3[t * 3 + j - 3];
        } else v = 0.f;
      } else {
        if (k < 67) {
          float vs = (k < 3) ? p.pos[s * 3 + k] : p.hN[s * 64 + k - 3];
          float vt = (k < 3) ? p.pos[t * 3 + k] : p.hN[t * 64 + k - 3];
          v = vs - vt;
        } else if (k < 134) {
          int j = k - 67;
          v = (j < 3) ? p.pos[t * 3 + j] : p.hN[t * 64 + j - 3];
        } else v = 0.f;
      }
      lds_m[w][i][k] = f2bf(v);
    }
  }
  __syncthreads();

  const int mrow = lane & 15;                 // M row / N col owned by this lane
  const int koff = (lane < 16) ? 0 : 8;

  // ---- GEMM1: [16 x KPAD] @ [KPAD x 64] ----
  v8f acc[4];
  #pragma unroll
  for (int nt = 0; nt < 4; ++nt)
    #pragma unroll
    for (int r = 0; r < 8; ++r) acc[nt][r] = 0.f;

  {
    const unsigned short* rowp = &lds_m[w][mrow][0];
    #pragma unroll
    for (int kt = 0; kt < KPAD / 32; ++kt) {
      AFrag a;
      const int kb = kt * 32 + koff;
      #pragma unroll
      for (int q = 0; q < 4; ++q) {
        a.u[q]     = *(const unsigned*)(rowp + kb + 2 * q);
        a.u[4 + q] = *(const unsigned*)(rowp + kb + 16 + 2 * q);
      }
      #pragma unroll
      for (int nt = 0; nt < 4; ++nt) {
        v16bf bv = *(const v16bf*)(p.W0p + ((kt * 4 + nt) * 512 + lane * 16));
        acc[nt] = __builtin_amdgcn_wmma_f32_16x16x32_bf16(
            false, a.v, false, bv, (short)0, acc[nt], false, false);
      }
    }
  }

  if (STAGE == 0) {
    // BN1 stats of h1 = relu(acc + b0)
    #pragma unroll
    for (int nt = 0; nt < 4; ++nt) {
      const int ch = nt * 16 + mrow;
      const float bias = p.b0[ch];
      float s = 0.f, q = 0.f;
      #pragma unroll
      for (int r = 0; r < 8; ++r) {
        float h = fmaxf(acc[nt][r] + bias, 0.f);
        s += h; q += h * h;
      }
      atomicAdd(&lds_stats[ch], s);
      atomicAdd(&lds_stats[64 + ch], q);
    }
    __syncthreads();
    atomicAdd(&p.stats[tid], lds_stats[tid]);
    return;
  }

  // ---- apply BN1, stash as bf16 A-tile for GEMM2 ----
  #pragma unroll
  for (int nt = 0; nt < 4; ++nt) {
    const int ch = nt * 16 + mrow;
    const float bias = p.b0[ch];
    const float a1 = p.bn1[ch], c1 = p.bn1[64 + ch];
    #pragma unroll
    for (int r = 0; r < 8; ++r) {
      const int m = (lane < 16) ? r : (8 + r);
      float h = fmaxf(acc[nt][r] + bias, 0.f);
      lds_m2[w][m][ch] = f2bf(h * a1 + c1);
    }
  }
  __syncthreads();

  // ---- GEMM2: [16 x 64] @ [64 x 64] ----
  v8f acc2[4];
  #pragma unroll
  for (int nt = 0; nt < 4; ++nt)
    #pragma unroll
    for (int r = 0; r < 8; ++r) acc2[nt][r] = 0.f;

  {
    const unsigned short* rowp2 = &lds_m2[w][mrow][0];
    #pragma unroll
    for (int kt = 0; kt < 2; ++kt) {
      AFrag a;
      const int kb = kt * 32 + koff;
      #pragma unroll
      for (int q = 0; q < 4; ++q) {
        a.u[q]     = *(const unsigned*)(rowp2 + kb + 2 * q);
        a.u[4 + q] = *(const unsigned*)(rowp2 + kb + 16 + 2 * q);
      }
      #pragma unroll
      for (int nt = 0; nt < 4; ++nt) {
        v16bf bv = *(const v16bf*)(p.W1p + ((kt * 4 + nt) * 512 + lane * 16));
        acc2[nt] = __builtin_amdgcn_wmma_f32_16x16x32_bf16(
            false, a.v, false, bv, (short)0, acc2[nt], false, false);
      }
    }
  }

  if (STAGE == 1) {
    // BN2 stats of h2 = relu(acc2 + b1)
    #pragma unroll
    for (int nt = 0; nt < 4; ++nt) {
      const int ch = nt * 16 + mrow;
      const float bias = p.b1[ch];
      float s = 0.f, q = 0.f;
      #pragma unroll
      for (int r = 0; r < 8; ++r) {
        float h = fmaxf(acc2[nt][r] + bias, 0.f);
        s += h; q += h * h;
      }
      atomicAdd(&lds_stats[ch], s);
      atomicAdd(&lds_stats[64 + ch], q);
    }
    __syncthreads();
    atomicAdd(&p.stats[128 + tid], lds_stats[tid]);
    return;
  }

  // ---- STAGE 2: apply BN2 and scatter-max to agg[dst] ----
  #pragma unroll
  for (int nt = 0; nt < 4; ++nt) {
    const int ch = nt * 16 + mrow;
    const float bias = p.b1[ch];
    const float a2 = p.bn2[ch], c2 = p.bn2[64 + ch];
    #pragma unroll
    for (int r = 0; r < 8; ++r) {
      const int m = (lane < 16) ? r : (8 + r);
      float h = fmaxf(acc2[nt][r] + bias, 0.f);
      float y = h * a2 + c2;
      int e = eb + m;
      int d = p.dst[e];
      atomicMaxFloat(&p.agg[d * 64 + ch], y);
    }
  }
}

// ---------------------------------------------------------------------------
// Head: jump-cat of 4 pooled means -> lin1 -> relu -> lin2 -> log_softmax
// ---------------------------------------------------------------------------
__global__ __launch_bounds__(512)
void head_kernel(const float* __restrict__ pooled,  // [4,B,64] sums
                 const float* __restrict__ cnt,     // [B]
                 const float* __restrict__ lin1_W, const float* __restrict__ lin1_b,
                 const float* __restrict__ lin2_W, const float* __restrict__ lin2_b,
                 float* __restrict__ out) {
  __shared__ float jf[B][4 * HID];
  __shared__ float z1[B][HID];
  __shared__ float z2[B][NCLS];
  const int t = threadIdx.x;

  for (int idx = t; idx < 4 * B * 64; idx += 512) {
    int l = idx / (B * 64);
    int r = idx % (B * 64);
    int b = r >> 6, c = r & 63;
    jf[b][l * 64 + c] = pooled[idx] / fmaxf(cnt[b], 1.0f);
  }
  __syncthreads();

  {  // z1 = relu(j @ lin1 + b): 8*64 = 512 threads
    int b = t >> 6, c = t & 63;
    float acc = lin1_b[c];
    #pragma unroll 8
    for (int k = 0; k < 4 * HID; ++k) acc += jf[b][k] * lin1_W[k * 64 + c];
    z1[b][c] = fmaxf(acc, 0.f);
  }
  __syncthreads();

  if (t < B * NCLS) {
    int b = t / NCLS, c = t % NCLS;
    float acc = lin2_b[c];
    #pragma unroll 8
    for (int k = 0; k < HID; ++k) acc += z1[b][k] * lin2_W[k * NCLS + c];
    z2[b][c] = acc;
  }
  __syncthreads();

  if (t < B * NCLS) {
    int b = t / NCLS, c = t % NCLS;
    float mx = -INFINITY;
    for (int i = 0; i < NCLS; ++i) mx = fmaxf(mx, z2[b][i]);
    float se = 0.f;
    for (int i = 0; i < NCLS; ++i) se += expf(z2[b][i] - mx);
    out[b * NCLS + c] = z2[b][c] - mx - logf(se);
  }
}

// ---------------------------------------------------------------------------
// Host orchestration
// ---------------------------------------------------------------------------
extern "C" void kernel_launch(void* const* d_in, const int* in_sizes, int n_in,
                              void* d_out, int out_size, void* d_ws, size_t ws_size,
                              hipStream_t stream) {
  const float* x3        = (const float*)d_in[0];
  const float* pos       = (const float*)d_in[1];
  const float* edge_attr = (const float*)d_in[2];
  const int*   eidx      = (const int*)d_in[3];
  const int*   batch     = (const int*)d_in[4];
  const float* c1_W0  = (const float*)d_in[5];
  const float* c1_b0  = (const float*)d_in[6];
  const float* c1_g0  = (const float*)d_in[7];
  const float* c1_be0 = (const float*)d_in[8];
  const float* c1_W1  = (const float*)d_in[9];
  const float* c1_b1  = (const float*)d_in[10];
  const float* c1_g1  = (const float*)d_in[11];
  const float* c1_be1 = (const float*)d_in[12];
  const float* cw_W0  = (const float*)d_in[13];
  const float* cw_b0  = (const float*)d_in[14];
  const float* cw_g0  = (const float*)d_in[15];
  const float* cw_be0 = (const float*)d_in[16];
  const float* cw_W1  = (const float*)d_in[17];
  const float* cw_b1  = (const float*)d_in[18];
  const float* cw_g1  = (const float*)d_in[19];
  const float* cw_be1 = (const float*)d_in[20];
  const float* lin1_W = (const float*)d_in[21];
  const float* lin1_b = (const float*)d_in[22];
  const float* lin2_W = (const float*)d_in[23];
  const float* lin2_b = (const float*)d_in[24];

  const int* src = eidx;
  const int* dst = eidx + E;

  // ---- carve workspace (256B aligned chunks) ----
  char* wsb = (char*)d_ws;
  size_t off = 0;
  auto carve = [&](size_t bytes) -> char* {
    char* r = wsb + off;
    off += (bytes + 255) & ~(size_t)255;
    return r;
  };
  float* hA     = (float*)carve((size_t)N * 64 * 4);
  float* agg    = (float*)carve((size_t)N * 64 * 4);
  float* pooled = (float*)carve((size_t)4 * B * 64 * 4);
  float* cnt    = (float*)carve((size_t)B * 4);
  float* stats  = (float*)carve(256 * 4);
  float* bn1    = (float*)carve(128 * 4);
  float* bn2    = (float*)carve(128 * 4);
  unsigned short* Wp_c10 = (unsigned short*)carve((size_t)32 * 64 * 2);
  unsigned short* Wp_c11 = (unsigned short*)carve((size_t)64 * 64 * 2);
  unsigned short* Wp_cw0[3];
  unsigned short* Wp_cw1[3];
  for (int l = 0; l < 3; ++l) {
    Wp_cw0[l] = (unsigned short*)carve((size_t)160 * 64 * 2);
    Wp_cw1[l] = (unsigned short*)carve((size_t)64 * 64 * 2);
  }
  (void)ws_size; (void)in_sizes; (void)n_in; (void)out_size;

  const float invE = 1.0f / (float)E;

  // ---- init pooling + counts ----
  fill_kernel<<<(B + 255) / 256, 256, 0, stream>>>(cnt, 0.f, B);
  fill_kernel<<<(4 * B * 64 + 255) / 256, 256, 0, stream>>>(pooled, 0.f, 4 * B * 64);
  count_kernel<<<(N + 255) / 256, 256, 0, stream>>>(batch, cnt);

  // ---- pack weights to WMMA B-fragment layout ----
  pack_w_kernel<<<(32 * 64 + 255) / 256, 256, 0, stream>>>(c1_W0, Wp_c10, 13, 32);
  pack_w_kernel<<<(64 * 64 + 255) / 256, 256, 0, stream>>>(c1_W1, Wp_c11, 64, 64);
  for (int l = 0; l < 3; ++l) {
    pack_w_kernel<<<(160 * 64 + 255) / 256, 256, 0, stream>>>(cw_W0 + (size_t)l * 134 * 64, Wp_cw0[l], 134, 160);
    pack_w_kernel<<<(64 * 64 + 255) / 256, 256, 0, stream>>>(cw_W1 + (size_t)l * 64 * 64, Wp_cw1[l], 64, 64);
  }

  // ---- conv1 ----
  {
    ConvParams cp;
    cp.pos = pos; cp.x3 = x3; cp.hN = nullptr; cp.edge_attr = edge_attr;
    cp.src = src; cp.dst = dst;
    cp.W0p = Wp_c10; cp.W1p = Wp_c11;
    cp.b0 = c1_b0; cp.b1 = c1_b1;
    cp.stats = stats; cp.bn1 = bn1; cp.bn2 = bn2; cp.agg = agg;

    fill_kernel<<<(256 + 255) / 256, 256, 0, stream>>>(stats, 0.f, 256);
    conv_stage_kernel<32, true, 0><<<EBLK, 128, 0, stream>>>(cp);
    bn_finalize_kernel<<<1, 64, 0, stream>>>(stats, c1_g0, c1_be0, bn1, invE);
    conv_stage_kernel<32, true, 1><<<EBLK, 128, 0, stream>>>(cp);
    bn_finalize_kernel<<<1, 64, 0, stream>>>(stats + 128, c1_g1, c1_be1, bn2, invE);
    fill_kernel<<<((size_t)N * 64 + 255) / 256, 256, 0, stream>>>(agg, -INFINITY, N * 64);
    conv_stage_kernel<32, true, 2><<<EBLK, 128, 0, stream>>>(cp);
    relu_pool_kernel<<<((size_t)N * 64 + 255) / 256, 256, 0, stream>>>(agg, hA, batch, pooled);
  }

  // ---- 3 x EdgeConvWeighted ----
  for (int l = 0; l < 3; ++l) {
    ConvParams cp;
    cp.pos = pos; cp.x3 = nullptr; cp.hN = hA; cp.edge_attr = nullptr;
    cp.src = src; cp.dst = dst;
    cp.W0p = Wp_cw0[l]; cp.W1p = Wp_cw1[l];
    cp.b0 = cw_b0 + (size_t)l * 64; cp.b1 = cw_b1 + (size_t)l * 64;
    cp.stats = stats; cp.bn1 = bn1; cp.bn2 = bn2; cp.agg = agg;

    fill_kernel<<<(256 + 255) / 256, 256, 0, stream>>>(stats, 0.f, 256);
    conv_stage_kernel<160, false, 0><<<EBLK, 128, 0, stream>>>(cp);
    bn_finalize_kernel<<<1, 64, 0, stream>>>(stats, cw_g0 + (size_t)l * 64, cw_be0 + (size_t)l * 64, bn1, invE);
    conv_stage_kernel<160, false, 1><<<EBLK, 128, 0, stream>>>(cp);
    bn_finalize_kernel<<<1, 64, 0, stream>>>(stats + 128, cw_g1 + (size_t)l * 64, cw_be1 + (size_t)l * 64, bn2, invE);
    fill_kernel<<<((size_t)N * 64 + 255) / 256, 256, 0, stream>>>(agg, -INFINITY, N * 64);
    conv_stage_kernel<160, false, 2><<<EBLK, 128, 0, stream>>>(cp);
    relu_pool_kernel<<<((size_t)N * 64 + 255) / 256, 256, 0, stream>>>(agg, hA, batch, pooled + (size_t)(l + 1) * B * 64);
  }

  // ---- head ----
  head_kernel<<<1, 512, 0, stream>>>(pooled, cnt, lin1_W, lin1_b, lin2_W, lin2_b, (float*)d_out);
}
